// SchNetWithDropout_72713796322231
// MI455X (gfx1250) — matrix operand order repaired and checked by reading, not compiled, not run on hardware
//
#include <hip/hip_runtime.h>
#include <hip/hip_bf16.h>

#define NATOMS 50000
#define NEDGES 1000000
#define ADIM 128
#define FDIM 128
#define RRBF 50
#define LLAYERS 3
#define CUTOFF_R 5.0f

typedef __attribute__((ext_vector_type(16))) _Float16 v16h;
typedef __attribute__((ext_vector_type(8)))  float    v8f;

// ---------------- WMMA fragment loaders (layouts per CDNA5 ISA §7.12.2) ---

// A-matrix 16x32 f16: lane holds row M=lane&15; elems 0..7 -> K = koff+0..7,
// elems 8..15 -> K = 16+koff+0..7, koff = (lane<16)?0:8.
__device__ __forceinline__ v16h wmma_load_a_f16(const _Float16* base, int ld, int lane) {
  int row  = lane & 15;
  int koff = (lane & 16) ? 8 : 0;
  const _Float16* p = base + row * ld + koff;
  v16h a;
#pragma unroll
  for (int e = 0; e < 8; ++e) a[e] = p[e];
#pragma unroll
  for (int e = 0; e < 8; ++e) a[e + 8] = p[16 + e];
  return a;
}

// A fragment built from an f32 row (on-the-fly f32->f16 convert)
__device__ __forceinline__ v16h wmma_load_a_f32(const float* rowp, int kbase, int lane) {
  int koff = kbase + ((lane & 16) ? 8 : 0);
  v16h a;
#pragma unroll
  for (int e = 0; e < 8; ++e) a[e] = (_Float16)rowp[koff + e];
#pragma unroll
  for (int e = 0; e < 8; ++e) a[e + 8] = (_Float16)rowp[koff + 16 + e];
  return a;
}

// B-matrix 32x16 f16 from row-major [K x N] memory: lane holds column N=lane&15,
// elems 0..7 -> K = koff+0..7, elems 8..15 -> K = 16+koff+0..7, koff = (lane<16)?0:8.
__device__ __forceinline__ v16h wmma_load_b_f16(const _Float16* base, int ld, int lane) {
  int koff = (lane & 16) ? 8 : 0;
  const _Float16* p = base + (lane & 15) + koff * ld;
  v16h b;
#pragma unroll
  for (int e = 0; e < 8; ++e) b[e] = p[e * ld];
#pragma unroll
  for (int e = 0; e < 8; ++e) b[e + 8] = p[(16 + e) * ld];
  return b;
}

__device__ __forceinline__ v8f wmma_f16(v16h a, v16h b, v8f c) {
  return __builtin_amdgcn_wmma_f32_16x16x32_f16(false, a, false, b, (short)0, c, false, false);
}

// shifted softplus: log(1+e^x) - log(2)
__device__ __forceinline__ float sspf(float x) {
  float sp = (x > 20.f) ? x : log1pf(__expf(x));
  return sp - 0.69314718056f;
}

// RBF A-fragment for one K-chunk c (32 K values), given distance d
__device__ __forceinline__ v16h rbf_frag(float d, int c, int koff) {
  const float width = CUTOFF_R / (float)(RRBF - 1);
  const float coeff = -0.5f / (width * width);
  v16h a;
#pragma unroll
  for (int e = 0; e < 16; ++e) {
    int k = c * 32 + ((e < 8) ? (koff + e) : (16 + koff + (e - 8)));
    float g = 0.f;
    if (k < RRBF) { float dt = d - (float)k * width; g = __expf(coeff * dt * dt); }
    a[e] = (_Float16)g;
  }
  return a;
}

// ---------------- small utility kernels ------------------------------------------

__global__ void k_embed(const int* __restrict__ Z, const float* __restrict__ emb,
                        float* __restrict__ x) {
  int i = blockIdx.x * blockDim.x + threadIdx.x;
  if (i < NATOMS * ADIM) x[i] = emb[Z[i >> 7] * ADIM + (i & 127)];
}

__global__ void k_zero(float* __restrict__ p, int n) {
  int i = blockIdx.x * blockDim.x + threadIdx.x;
  if (i < n) p[i] = 0.f;
}

__global__ void k_cvt(const float* __restrict__ s, _Float16* __restrict__ d, int n) {
  int i = blockIdx.x * blockDim.x + threadIdx.x;
  if (i < n) d[i] = (_Float16)s[i];
}

// fw1 [L,50,128] -> padded [L,64,128] f16 (rows 50..63 zero)
__global__ void k_cvt_fw1(const float* __restrict__ s, _Float16* __restrict__ d) {
  int i = blockIdx.x * blockDim.x + threadIdx.x;
  if (i < LLAYERS * 64 * 128) {
    int l = i / (64 * 128);
    int r = (i >> 7) & 63;
    int c = i & 127;
    d[i] = (r < RRBF) ? (_Float16)s[(l * RRBF + r) * 128 + c] : (_Float16)0.f;
  }
}

// ---------------- h = x @ W_in2f  (f32 in, f16 W, f16 out) ------------------------
// wave -> 32 rows x 128 cols (two A row-blocks share each B fragment);
// block = 4 waves = 128 rows
__global__ __launch_bounds__(128) void k_gemm_h(const float* __restrict__ x,
                                                const _Float16* __restrict__ W,
                                                _Float16* __restrict__ h) {
  int lane = threadIdx.x & 31;
  int wave = threadIdx.x >> 5;
  int rowbase = (blockIdx.x * 4 + wave) * 32;

  const float* xr0;
  const float* xr1;
  {
    int r0 = rowbase + (lane & 15);       if (r0 >= NATOMS) r0 = NATOMS - 1;
    int r1 = rowbase + 16 + (lane & 15);  if (r1 >= NATOMS) r1 = NATOMS - 1;
    xr0 = x + (size_t)r0 * ADIM;
    xr1 = x + (size_t)r1 * ADIM;
  }

  v8f acc[2][8] = {};
#pragma unroll
  for (int c = 0; c < 4; ++c) {
    v16h a0 = wmma_load_a_f32(xr0, c * 32, lane);
    v16h a1 = wmma_load_a_f32(xr1, c * 32, lane);
#pragma unroll
    for (int t = 0; t < 8; ++t) {
      v16h b = wmma_load_b_f16(W + (c * 32) * FDIM + t * 16, FDIM, lane);
      acc[0][t] = wmma_f16(a0, b, acc[0][t]);
      acc[1][t] = wmma_f16(a1, b, acc[1][t]);
    }
  }
  int hi = (lane & 16) ? 8 : 0;
  int n0 = lane & 15;
#pragma unroll
  for (int r = 0; r < 2; ++r) {
#pragma unroll
    for (int t = 0; t < 8; ++t) {
#pragma unroll
      for (int v = 0; v < 8; ++v) {
        int m = rowbase + r * 16 + hi + v;
        if (m < NATOMS) h[(size_t)m * FDIM + t * 16 + n0] = (_Float16)acc[r][t][v];
      }
    }
  }
}

// ---------------- fused edge kernel -----------------------------------------------
// wave -> 32 edges: RBF -> GEMM(fw1) -> ssp -> LDS relayout -> GEMM(fw2)
//   -> *rcut -> gather h[idx_j] -> atomicAdd agg[idx_i]
// block = 4 waves = 128 edges
__global__ __launch_bounds__(128) void k_edge(const float* __restrict__ r_ij,
                                              const int* __restrict__ idx_i,
                                              const int* __restrict__ idx_j,
                                              const _Float16* __restrict__ fw1h,
                                              const float* __restrict__ fb1,
                                              const _Float16* __restrict__ fw2h,
                                              const float* __restrict__ fb2,
                                              const _Float16* __restrict__ h,
                                              float* __restrict__ agg) {
  __shared__ _Float16 s_inter[4][32 * FDIM];
  __shared__ float    s_rcut[4][32];
  __shared__ int      s_ii[4][32];
  __shared__ int      s_jj[4][32];

  int lane = threadIdx.x & 31;
  int wave = threadIdx.x >> 5;
  int ebase = (blockIdx.x * 4 + wave) * 32;

  int m0 = lane & 15;
  // distances for both row-blocks this lane feeds
  int e0 = ebase + m0;       int ec0 = (e0 < NEDGES) ? e0 : (NEDGES - 1);
  int e1 = ebase + 16 + m0;  int ec1 = (e1 < NEDGES) ? e1 : (NEDGES - 1);
  float d0, d1;
  {
    float rx = r_ij[ec0 * 3 + 0], ry = r_ij[ec0 * 3 + 1], rz = r_ij[ec0 * 3 + 2];
    d0 = sqrtf(rx * rx + ry * ry + rz * rz);
    rx = r_ij[ec1 * 3 + 0]; ry = r_ij[ec1 * 3 + 1]; rz = r_ij[ec1 * 3 + 2];
    d1 = sqrtf(rx * rx + ry * ry + rz * rz);
  }
  // per-lane table entry: edge (ebase+lane) -> rcut/idx (all 32 lanes, no divergence)
  {
    int me = ebase + lane;
    int mec = (me < NEDGES) ? me : (NEDGES - 1);
    float dm = (lane & 16) ? d1 : d0;
    float rc = (dm < CUTOFF_R)
                 ? 0.5f * (__cosf(dm * (3.14159265358979f / CUTOFF_R)) + 1.0f) : 0.f;
    s_rcut[wave][lane] = rc;
    s_ii[wave][lane]   = idx_i[mec];
    s_jj[wave][lane]   = idx_j[mec];
  }

  // GEMM1: rbf[32x64] @ fw1h[64x128]
  int koff = (lane & 16) ? 8 : 0;
  v8f acc[2][8] = {};
#pragma unroll
  for (int c = 0; c < 2; ++c) {
    v16h a0 = rbf_frag(d0, c, koff);
    v16h a1 = rbf_frag(d1, c, koff);
#pragma unroll
    for (int t = 0; t < 8; ++t) {
      v16h b = wmma_load_b_f16(fw1h + (c * 32) * FDIM + t * 16, FDIM, lane);
      acc[0][t] = wmma_f16(a0, b, acc[0][t]);
      acc[1][t] = wmma_f16(a1, b, acc[1][t]);
    }
  }

  // bias + ssp, stash into LDS row-major for A-relayout
  int hi = (lane & 16) ? 8 : 0;
#pragma unroll
  for (int r = 0; r < 2; ++r) {
#pragma unroll
    for (int t = 0; t < 8; ++t) {
      float bn = fb1[t * 16 + m0];
#pragma unroll
      for (int v = 0; v < 8; ++v) {
        s_inter[wave][(r * 16 + hi + v) * FDIM + t * 16 + m0] =
            (_Float16)sspf(acc[r][t][v] + bn);
      }
    }
  }
  __syncthreads();

  // GEMM2: inter[32x128] @ fw2h[128x128]
  v8f acc2[2][8] = {};
#pragma unroll
  for (int c = 0; c < 4; ++c) {
    v16h a0 = wmma_load_a_f16(&s_inter[wave][c * 32], FDIM, lane);
    v16h a1 = wmma_load_a_f16(&s_inter[wave][16 * FDIM + c * 32], FDIM, lane);
#pragma unroll
    for (int t = 0; t < 8; ++t) {
      v16h b = wmma_load_b_f16(fw2h + (c * 32) * FDIM + t * 16, FDIM, lane);
      acc2[0][t] = wmma_f16(a0, b, acc2[0][t]);
      acc2[1][t] = wmma_f16(a1, b, acc2[1][t]);
    }
  }

  // cutoff modulate, gather h[idx_j], scatter-add agg[idx_i]
#pragma unroll
  for (int r = 0; r < 2; ++r) {
    int ii[8], jj[8];
    float rcv[8];
#pragma unroll
    for (int v = 0; v < 8; ++v) {
      int m = r * 16 + hi + v;
      ii[v]  = s_ii[wave][m];
      jj[v]  = s_jj[wave][m];
      rcv[v] = s_rcut[wave][m];
    }
#pragma unroll
    for (int t = 0; t < 8; ++t) {
      float bn = fb2[t * 16 + m0];
      int col = t * 16 + m0;
#pragma unroll
      for (int v = 0; v < 8; ++v) {
        int eg = ebase + r * 16 + hi + v;
        if (eg < NEDGES) {
          float wf = (acc2[r][t][v] + bn) * rcv[v];
          float hv = (float)h[(size_t)jj[v] * FDIM + col];
          atomicAdd(&agg[(size_t)ii[v] * FDIM + col], hv * wf);
        }
      }
    }
  }
}

// ---------------- output MLP + residual -------------------------------------------
// x += ssp(agg @ ow1 + ob1) @ ow2 + ob2 ; wave -> 32 atoms, block = 4 waves
__global__ __launch_bounds__(128) void k_outmlp(const float* __restrict__ agg,
                                                const _Float16* __restrict__ ow1h,
                                                const float* __restrict__ ob1,
                                                const _Float16* __restrict__ ow2h,
                                                const float* __restrict__ ob2,
                                                float* __restrict__ x) {
  __shared__ _Float16 s_inter[4][32 * ADIM];

  int lane = threadIdx.x & 31;
  int wave = threadIdx.x >> 5;
  int rowbase = (blockIdx.x * 4 + wave) * 32;
  int m0 = lane & 15;

  const float* ar0;
  const float* ar1;
  {
    int r0 = rowbase + m0;       if (r0 >= NATOMS) r0 = NATOMS - 1;
    int r1 = rowbase + 16 + m0;  if (r1 >= NATOMS) r1 = NATOMS - 1;
    ar0 = agg + (size_t)r0 * FDIM;
    ar1 = agg + (size_t)r1 * FDIM;
  }

  v8f acc[2][8] = {};
#pragma unroll
  for (int c = 0; c < 4; ++c) {
    v16h a0 = wmma_load_a_f32(ar0, c * 32, lane);
    v16h a1 = wmma_load_a_f32(ar1, c * 32, lane);
#pragma unroll
    for (int t = 0; t < 8; ++t) {
      v16h b = wmma_load_b_f16(ow1h + (c * 32) * ADIM + t * 16, ADIM, lane);
      acc[0][t] = wmma_f16(a0, b, acc[0][t]);
      acc[1][t] = wmma_f16(a1, b, acc[1][t]);
    }
  }

  int hi = (lane & 16) ? 8 : 0;
#pragma unroll
  for (int r = 0; r < 2; ++r) {
#pragma unroll
    for (int t = 0; t < 8; ++t) {
      float bn = ob1[t * 16 + m0];
#pragma unroll
      for (int v = 0; v < 8; ++v) {
        s_inter[wave][(r * 16 + hi + v) * ADIM + t * 16 + m0] =
            (_Float16)sspf(acc[r][t][v] + bn);
      }
    }
  }
  __syncthreads();

  v8f acc2[2][8] = {};
#pragma unroll
  for (int c = 0; c < 4; ++c) {
    v16h a0 = wmma_load_a_f16(&s_inter[wave][c * 32], ADIM, lane);
    v16h a1 = wmma_load_a_f16(&s_inter[wave][16 * ADIM + c * 32], ADIM, lane);
#pragma unroll
    for (int t = 0; t < 8; ++t) {
      v16h b = wmma_load_b_f16(ow2h + (c * 32) * ADIM + t * 16, ADIM, lane);
      acc2[0][t] = wmma_f16(a0, b, acc2[0][t]);
      acc2[1][t] = wmma_f16(a1, b, acc2[1][t]);
    }
  }

#pragma unroll
  for (int r = 0; r < 2; ++r) {
#pragma unroll
    for (int t = 0; t < 8; ++t) {
      float bn = ob2[t * 16 + m0];
#pragma unroll
      for (int v = 0; v < 8; ++v) {
        int m = rowbase + r * 16 + hi + v;
        if (m < NATOMS) x[(size_t)m * ADIM + t * 16 + m0] += acc2[r][t][v] + bn;
      }
    }
  }
}

// ---------------- host orchestration ----------------------------------------------

extern "C" void kernel_launch(void* const* d_in, const int* in_sizes, int n_in,
                              void* d_out, int out_size, void* d_ws, size_t ws_size,
                              hipStream_t stream) {
  const int*   Z         = (const int*)d_in[0];
  const float* r_ij      = (const float*)d_in[1];
  const int*   idx_i     = (const int*)d_in[2];
  const int*   idx_j     = (const int*)d_in[3];
  const float* embedding = (const float*)d_in[4];
  const float* W_in2f    = (const float*)d_in[5];
  const float* fw1       = (const float*)d_in[6];
  const float* fb1       = (const float*)d_in[7];
  const float* fw2       = (const float*)d_in[8];
  const float* fb2       = (const float*)d_in[9];
  const float* ow1       = (const float*)d_in[10];
  const float* ob1       = (const float*)d_in[11];
  const float* ow2       = (const float*)d_in[12];
  const float* ob2       = (const float*)d_in[13];

  float* x = (float*)d_out;  // [N,128] lives directly in d_out

  char* ws = (char*)d_ws;
  size_t off = 0;
  _Float16* h16 = (_Float16*)(ws + off); off += (size_t)NATOMS * FDIM * sizeof(_Float16);
  float* agg    = (float*)(ws + off);    off += (size_t)NATOMS * FDIM * sizeof(float);
  _Float16* w_in2f_h = (_Float16*)(ws + off); off += (size_t)LLAYERS * 128 * 128 * 2;
  _Float16* fw1_h    = (_Float16*)(ws + off); off += (size_t)LLAYERS * 64 * 128 * 2;
  _Float16* fw2_h    = (_Float16*)(ws + off); off += (size_t)LLAYERS * 128 * 128 * 2;
  _Float16* ow1_h    = (_Float16*)(ws + off); off += (size_t)LLAYERS * 128 * 128 * 2;
  _Float16* ow2_h    = (_Float16*)(ws + off); off += (size_t)LLAYERS * 128 * 128 * 2;

  // x = embedding[Z]
  k_embed<<<(NATOMS * ADIM + 255) / 256, 256, 0, stream>>>(Z, embedding, x);

  // weight conversions to f16 (cheap, deterministic)
  int nw = LLAYERS * 128 * 128;
  k_cvt<<<(nw + 255) / 256, 256, 0, stream>>>(W_in2f, w_in2f_h, nw);
  k_cvt_fw1<<<(LLAYERS * 64 * 128 + 255) / 256, 256, 0, stream>>>(fw1, fw1_h);
  k_cvt<<<(nw + 255) / 256, 256, 0, stream>>>(fw2, fw2_h, nw);
  k_cvt<<<(nw + 255) / 256, 256, 0, stream>>>(ow1, ow1_h, nw);
  k_cvt<<<(nw + 255) / 256, 256, 0, stream>>>(ow2, ow2_h, nw);

  for (int l = 0; l < LLAYERS; ++l) {
    k_zero<<<(NATOMS * FDIM + 255) / 256, 256, 0, stream>>>(agg, NATOMS * FDIM);
    k_gemm_h<<<(NATOMS + 127) / 128, 128, 0, stream>>>(x, w_in2f_h + (size_t)l * 128 * 128, h16);
    k_edge<<<(NEDGES + 127) / 128, 128, 0, stream>>>(r_ij, idx_i, idx_j,
        fw1_h + (size_t)l * 64 * 128, fb1 + l * 128,
        fw2_h + (size_t)l * 128 * 128, fb2 + l * 128, h16, agg);
    k_outmlp<<<(NATOMS + 127) / 128, 128, 0, stream>>>(agg,
        ow1_h + (size_t)l * 128 * 128, ob1 + l * 128,
        ow2_h + (size_t)l * 128 * 128, ob2 + l * 128, x);
  }
}
